// SwitchingGRU1_18940805775602
// MI455X (gfx1250) — matrix operand order repaired and checked by reading, not compiled, not run on hardware
//
#include <hip/hip_runtime.h>
#include <hip/hip_bf16.h>

typedef __attribute__((ext_vector_type(16))) __bf16 v16bf;
typedef __attribute__((ext_vector_type(8)))  float  v8f;

// Explicit global-address-space pointers so the opaque (asm-laundered) W
// stream still lowers to GLOBAL_LOAD_B128 (LOADcnt only) instead of
// FLAT_LOAD_B128 (LOADcnt+DScnt, shares the LDS path with the WMMA B reads).
typedef __attribute__((address_space(1))) const __bf16* gptr_bf16;
typedef __attribute__((address_space(1))) const v16bf*  gptr_v16bf;

// Problem constants
#define Bq   256
#define Tq   512
#define Hd   128
#define Sd   8
#define NSW  6
#define NCOL 16          // batch columns per workgroup
#define NWAVE 18         // 3 gates * 6 switches
#define NTHR (NWAVE*32)  // 576 threads

// ---- dynamic LDS byte offsets -------------------------------------------
#define YLDS_OFF   0u        // 18*16*128 f32 = 147456 B  (Y[g*6+s][col][m])
#define HF_OFF     147456u   // 16*128 f32    = 8192 B
#define HBF_OFF    155648u   // 16*128 bf16   = 4096 B   (32B aligned)
#define XL_OFF     159744u   // 16*384 f32    = 24576 B
#define WINL_OFF   184320u   // 384*8 f32     = 12288 B
#define BINL_OFF   196608u   // 384 f32       = 1536 B
#define BHR_OFF    198144u   // 6*128 f32     = 3072 B
#define BHZ_OFF    201216u
#define BHN_OFF    204288u
#define WOL_OFF    207360u   // 2*128 f32 (padded)
#define BOL_OFF    208384u   // 2 f32
#define STIM_OFF   208392u   // 16*8 f32
#define SID_OFF    208904u   // 16 i32
#define MASK_OFF   208968u   // 16 f32
#define SMEM_BYTES 209032u

// -------------------------------------------------------------------------
// Pre-kernel: convert W_hr/W_hz/W_hn (f32, 6 x 128 x 128 each) into bf16,
// pre-swizzled into WMMA A fragments (16x32 bf16 tile, contiguous k = 16*(lane/16)+e, not the
// ISA 7.12.2 table order): fragment index p = (((mt*4+kt)*32)+lane)*16 + e.
// -------------------------------------------------------------------------
__global__ void gru_convert_w(const float* __restrict__ Wr,
                              const float* __restrict__ Wz,
                              const float* __restrict__ Wn,
                              __bf16* __restrict__ outw) {
  const int m = blockIdx.x;          // 0..17
  const int g = m / 6, s = m - g * 6;
  const float* W = (g == 0 ? Wr : (g == 1 ? Wz : Wn)) + s * (Hd * Hd);
  __bf16* o = outw + m * (Hd * Hd);
  for (int p = threadIdx.x; p < Hd * Hd; p += blockDim.x) {
    const int e    = p & 15;
    const int lane = (p >> 4) & 31;
    const int kt   = (p >> 9) & 3;
    const int mt   = (p >> 11) & 7;
    int row, k;
    if (lane < 16) { row = mt * 16 + lane;      k = kt * 32 + e;  }
    else           { row = mt * 16 + lane - 16; k = kt * 32 + 16 + e; }
    o[p] = (__bf16)W[row * Hd + k];
  }
}

// -------------------------------------------------------------------------
// Main kernel: one workgroup = 16 batch columns, 18 waves. Per time step:
//  A: stage stim/sid/mask tile
//  B: all threads compute x = stim@Win^T + bin; each wave runs a dense
//     128x16x128 bf16 WMMA GEMM Y[g][s] = W[g][s] @ H for its (gate,switch),
//     double-buffer-pipelining the pre-swizzled W fragment stream from the
//     L2-resident weight array (re-fetching each step beats spilling 256
//     VGPRs of loop-invariant fragments to scratch).
//  C: per-element select by sid + GRU gating -> new H (f32 + bf16 copies)
//  D: 2-wide output projection -> logits
// -------------------------------------------------------------------------
__global__ void __launch_bounds__(NTHR) gru_scan(
    const float* __restrict__ stim, const int* __restrict__ swid,
    const float* __restrict__ maskg,
    const float* __restrict__ Win,  const float* __restrict__ binp,
    const float* __restrict__ bhr,  const float* __restrict__ bhz,
    const float* __restrict__ bhn,
    const float* __restrict__ Wo,   const float* __restrict__ bo,
    const __bf16* __restrict__ Wbf, float* __restrict__ out) {
  extern __shared__ char smem[];
  float*  Ylds  = (float*)(smem + YLDS_OFF);
  float*  Hf    = (float*)(smem + HF_OFF);
  __bf16* Hbf   = (__bf16*)(smem + HBF_OFF);
  float*  Xl    = (float*)(smem + XL_OFF);
  float*  WinL  = (float*)(smem + WINL_OFF);
  float*  binL  = (float*)(smem + BINL_OFF);
  float*  bhrL  = (float*)(smem + BHR_OFF);
  float*  bhzL  = (float*)(smem + BHZ_OFF);
  float*  bhnL  = (float*)(smem + BHN_OFF);
  float*  WoL   = (float*)(smem + WOL_OFF);
  float*  boL   = (float*)(smem + BOL_OFF);
  float*  stimT = (float*)(smem + STIM_OFF);
  int*    sidT  = (int*)(smem + SID_OFF);
  float*  maskT = (float*)(smem + MASK_OFF);

  const int tid  = threadIdx.x;
  const int lane = tid & 31;
  const int wave = tid >> 5;                 // 0..17 == g*6+s
  const int colBase = blockIdx.x * NCOL;

  // one-time staging of small operands + zero hidden state
  for (int i = tid; i < 3 * Hd * Sd; i += NTHR) WinL[i] = Win[i];
  for (int i = tid; i < 3 * Hd;      i += NTHR) binL[i] = binp[i];
  for (int i = tid; i < NSW * Hd;    i += NTHR) { bhrL[i] = bhr[i]; bhzL[i] = bhz[i]; bhnL[i] = bhn[i]; }
  for (int i = tid; i < 2 * Hd;      i += NTHR) WoL[i] = Wo[i];
  if (tid < 2) boL[tid] = bo[tid];
  for (int i = tid; i < NCOL * Hd;   i += NTHR) { Hf[i] = 0.0f; Hbf[i] = (__bf16)0.0f; }
  __syncthreads();

  // Per-wave weight-fragment base, typed as an addrspace(1) pointer and made
  // opaque inside the t-loop (held in a VGPR pair; it is threadIdx-derived,
  // hence divergent to the compiler). This forces the fragments to be
  // streamed from L2 every step via global_load_b128 instead of being
  // hoisted + spilled to scratch.
  gptr_bf16 Wdyn = (gptr_bf16)(Wbf + wave * (Hd * Hd));
  const int bcol = lane & 15;
  const int bsel = (lane >> 4) & 1;

  for (int t = 0; t < Tq; ++t) {
    // ---- phase A: stage per-step tiles ----
    if (tid < NCOL * Sd) {
      const int col = tid >> 3, ss = tid & 7;
      stimT[tid] = stim[(size_t)(colBase + col) * (Tq * Sd) + t * Sd + ss];
    }
    if (tid < NCOL) {
      int v = swid[(colBase + tid) * Tq + t];
      sidT[tid]  = v < 0 ? 0 : (v > NSW - 1 ? NSW - 1 : v);
      maskT[tid] = maskg[(colBase + tid) * Tq + t];
    }
    __syncthreads();

    // ---- phase B(1): x = stim @ Win^T + bin  (16 x 384) ----
    for (int idx = tid; idx < NCOL * 3 * Hd; idx += NTHR) {
      const int col = idx / (3 * Hd), o = idx - col * (3 * Hd);
      float a = binL[o];
#pragma unroll
      for (int q = 0; q < Sd; ++q) a += stimT[col * Sd + q] * WinL[o * Sd + q];
      Xl[idx] = a;
    }

    // ---- phase B(2): dense WMMA GEMM  Y[wave] = W[wave] @ H  (128x16) ----
    // Two halves of 4 M-tiles; within each half the kt-loop is software-
    // pipelined: region kt issues the loads for kt+1, then runs WMMAs on the
    // fragments loaded in region kt-1 (wait becomes loadcnt<=4, not 0).
    asm volatile("" : "+v"(Wdyn));   // defeat LICM: force per-step W streaming
#pragma unroll
    for (int half = 0; half < 2; ++half) {
      v8f acc[4];
#pragma unroll
      for (int q = 0; q < 4; ++q) acc[q] = (v8f){0, 0, 0, 0, 0, 0, 0, 0};
      v16bf cur[4];
#pragma unroll
      for (int q = 0; q < 4; ++q)
        cur[q] = *(gptr_v16bf)(Wdyn + ((((half * 4 + q) * 4 + 0) * 32) + lane) * 16);
#pragma unroll
      for (int kt = 0; kt < 4; ++kt) {
        const v16bf bfrag = *(const v16bf*)(Hbf + bcol * Hd + kt * 32 + bsel * 16);
        v16bf nxt[4];
        if (kt < 3) {
#pragma unroll
          for (int q = 0; q < 4; ++q)
            nxt[q] = *(gptr_v16bf)(Wdyn + ((((half * 4 + q) * 4 + kt + 1) * 32) + lane) * 16);
        }
#pragma unroll
        for (int q = 0; q < 4; ++q)
          acc[q] = __builtin_amdgcn_wmma_f32_16x16x32_bf16(
              false, cur[q], false, bfrag, (short)0, acc[q], false, false);
        __builtin_amdgcn_sched_barrier(0);   // cap in-flight fragments
        if (kt < 3) {
#pragma unroll
          for (int q = 0; q < 4; ++q) cur[q] = nxt[q];
        }
      }
      {
        float* ybase = Ylds + (wave * NCOL + bcol) * Hd + bsel * 8;
#pragma unroll
        for (int q = 0; q < 4; ++q)
#pragma unroll
          for (int v = 0; v < 8; ++v) ybase[(half * 4 + q) * 16 + v] = acc[q][v];
      }
    }
    __syncthreads();

    // ---- phase C: switch-select + GRU gating ----
    for (int idx = tid; idx < NCOL * Hd; idx += NTHR) {
      const int col = idx >> 7, i = idx & (Hd - 1);
      const int sd  = sidT[col];
      const float h  = Hf[idx];
      const float yr = Ylds[((0 * NSW + sd) * NCOL + col) * Hd + i] + bhrL[sd * Hd + i];
      const float yz = Ylds[((1 * NSW + sd) * NCOL + col) * Hd + i] + bhzL[sd * Hd + i];
      const float yn = Ylds[((2 * NSW + sd) * NCOL + col) * Hd + i] + bhnL[sd * Hd + i];
      const float xr = Xl[col * 384 + i];
      const float xz = Xl[col * 384 + Hd + i];
      const float xn = Xl[col * 384 + 2 * Hd + i];
      const float r = 1.0f / (1.0f + expf(-(xr + yr)));
      const float z = 1.0f / (1.0f + expf(-(xz + yz)));
      const float n = tanhf(xn + r * yn);
      const float hnew = (1.0f - z) * n + z * h;
      const float mv = maskT[col];
      const float hout = mv * hnew + (1.0f - mv) * h;
      Hf[idx]  = hout;
      Hbf[idx] = (__bf16)hout;
    }
    __syncthreads();

    // ---- phase D: logits = h_out @ Wo^T + bo ----
    if (tid < NCOL * 2) {
      const int col = tid >> 1, j = tid & 1;
      float a = boL[j];
      const float* hp = Hf + col * Hd;
      const float* wp = WoL + j * Hd;
#pragma unroll 4
      for (int i = 0; i < Hd; ++i) a += hp[i] * wp[i];
      out[(size_t)(colBase + col) * (Tq * 2) + t * 2 + j] = a;
    }
    __syncthreads();
  }
}

extern "C" void kernel_launch(void* const* d_in, const int* in_sizes, int n_in,
                              void* d_out, int out_size, void* d_ws, size_t ws_size,
                              hipStream_t stream) {
  (void)in_sizes; (void)n_in; (void)out_size; (void)ws_size;
  const float* stim = (const float*)d_in[0];
  const int*   swid = (const int*)d_in[1];
  const float* mask = (const float*)d_in[2];
  const float* Win  = (const float*)d_in[3];
  const float* binp = (const float*)d_in[4];
  const float* Whr  = (const float*)d_in[5];
  const float* Whz  = (const float*)d_in[6];
  const float* Whn  = (const float*)d_in[7];
  const float* bhr  = (const float*)d_in[8];
  const float* bhz  = (const float*)d_in[9];
  const float* bhn  = (const float*)d_in[10];
  const float* Wo   = (const float*)d_in[11];
  const float* bo   = (const float*)d_in[12];
  float*  out = (float*)d_out;
  __bf16* Wbf = (__bf16*)d_ws;   // needs 18*128*128*2 = 589,824 bytes

  hipLaunchKernelGGL(gru_convert_w, dim3(NWAVE), dim3(256), 0, stream,
                     Whr, Whz, Whn, Wbf);

  (void)hipFuncSetAttribute(reinterpret_cast<const void*>(gru_scan),
                            hipFuncAttributeMaxDynamicSharedMemorySize,
                            (int)SMEM_BYTES);
  hipLaunchKernelGGL(gru_scan, dim3(Bq / NCOL), dim3(NTHR), SMEM_BYTES, stream,
                     stim, swid, mask, Win, binp, bhr, bhz, bhn, Wo, bo, Wbf, out);
}
